// PointCloudAutoencoder_11862699672003
// MI455X (gfx1250) — matrix operand order repaired and checked by reading, not compiled, not run on hardware
//
#include <hip/hip_runtime.h>
#include <hip/hip_bf16.h>
#include <cstdint>
#include <cstddef>

// ---------------------------------------------------------------------------
// Types for CDNA5 WMMA
// ---------------------------------------------------------------------------
typedef __attribute__((ext_vector_type(16))) _Float16 v16h;
typedef __attribute__((ext_vector_type(8)))  _Float16 v8h;
typedef __attribute__((ext_vector_type(8)))  float    v8f;

#define B_    8
#define N_    2048
#define KNN_  20
#define BN_   (B_ * N_)        // 16384
#define BNK_  (BN_ * KNN_)     // 327680

// ---------------------------------------------------------------------------
// f32 -> f16 conversion (point clouds)
// ---------------------------------------------------------------------------
__global__ void f32_to_f16_kernel(const float* __restrict__ src,
                                  _Float16* __restrict__ dst, size_t n) {
  size_t i = (size_t)blockIdx.x * blockDim.x + threadIdx.x;
  if (i < n) dst[i] = (_Float16)src[i];
}

// ---------------------------------------------------------------------------
// Weight conversion with transpose + K padding:
//   src: K x Nn (row-major f32)  ->  dst: Nn x Kp (row-major f16), zero-padded K
// ---------------------------------------------------------------------------
__global__ void transpose_f32_to_f16_kernel(const float* __restrict__ src,
                                            _Float16* __restrict__ dst,
                                            int K, int Nn, int Kp) {
  size_t gid = (size_t)blockIdx.x * blockDim.x + threadIdx.x;
  size_t total = (size_t)Nn * Kp;
  if (gid >= total) return;
  int k = (int)(gid % Kp);
  int n = (int)(gid / Kp);
  dst[gid] = (k < K) ? (_Float16)src[(size_t)k * Nn + n] : (_Float16)0.0f;
}

// ---------------------------------------------------------------------------
// KNN: per query point, scan all N candidates, keep 20 smallest squared dists.
// ---------------------------------------------------------------------------
template <int C>
__global__ void knn_kernel(const _Float16* __restrict__ X, int* __restrict__ idx) {
  int gid = blockIdx.x * blockDim.x + threadIdx.x;
  if (gid >= BN_) return;
  int b = gid / N_;
  int n = gid % N_;
  const _Float16* xi = X + ((size_t)b * N_ + n) * C;
  float q[C];
#pragma unroll
  for (int c = 0; c < C; ++c) q[c] = (float)xi[c];

  float bestd[KNN_];
  int   besti[KNN_];
#pragma unroll
  for (int k = 0; k < KNN_; ++k) { bestd[k] = 1e30f; besti[k] = 0; }

  const _Float16* Xb = X + (size_t)b * N_ * C;
  for (int m = 0; m < N_; ++m) {
    const _Float16* xm = Xb + (size_t)m * C;
    float d = 0.f;
    if (C % 8 == 0) {
      const v8h* xv = (const v8h*)xm;
#pragma unroll
      for (int c8 = 0; c8 < C / 8; ++c8) {
        v8h v = xv[c8];
#pragma unroll
        for (int j = 0; j < 8; ++j) {
          float t = q[c8 * 8 + j] - (float)v[j];
          d += t * t;
        }
      }
    } else {
#pragma unroll
      for (int c = 0; c < C; ++c) { float t = q[c] - (float)xm[c]; d += t * t; }
    }
    if (d < bestd[KNN_ - 1]) {
      int p = KNN_ - 1;
      while (p > 0 && bestd[p - 1] > d) {
        bestd[p] = bestd[p - 1]; besti[p] = besti[p - 1]; --p;
      }
      bestd[p] = d; besti[p] = m;
    }
  }
  int* out = idx + (size_t)gid * KNN_;
#pragma unroll
  for (int k = 0; k < KNN_; ++k) out[k] = besti[k];
}

// ---------------------------------------------------------------------------
// Edge feature gather with padded row stride S (>= 2C, multiple of 32):
//   G[p, 0:C] = x[j]-x[n], G[p, C:2C] = x[n], G[p, 2C:S] = 0
// ---------------------------------------------------------------------------
__global__ void edge_kernel(const _Float16* __restrict__ X,
                            const int* __restrict__ idx,
                            _Float16* __restrict__ G, int C, int S) {
  size_t total = (size_t)BNK_ * S;
  size_t gid = (size_t)blockIdx.x * blockDim.x + threadIdx.x;
  if (gid >= total) return;
  int c = (int)(gid % S);
  size_t t = gid / S;
  int k = (int)(t % KNN_); t /= KNN_;
  int n = (int)(t % N_);
  int b = (int)(t / N_);
  _Float16* g = G + (((size_t)b * N_ + n) * KNN_ + k) * (size_t)S;
  if (c < 2 * C) {
    int j = idx[((size_t)b * N_ + n) * KNN_ + k];
    if (c < C) {
      float vi = (float)X[((size_t)b * N_ + n) * C + c];
      float vj = (float)X[((size_t)b * N_ + j) * C + c];
      g[c] = (_Float16)(vj - vi);
    } else {
      g[c] = X[((size_t)b * N_ + n) * C + (c - C)];
    }
  } else {
    g[c] = (_Float16)0.0f;
  }
}

// ---------------------------------------------------------------------------
// WMMA GEMM:  C[M,Nn] = act(A[M,Kd] @ W[Kd,Nn] + bias)
//   A:  row-major f16, Kd multiple of 32, rows 0..M-1 backed (M mult of 16)
//   Wt: TRANSPOSED weights, Nn x Kd row-major f16
// Block = 8 waves: 128 rows x 64 cols of C.  B tile (32x64 per K-step) is
// DOUBLE-BUFFERED in LDS via GLOBAL_LOAD_ASYNC_TO_LDS_B128 (ASYNCcnt): the
// copy of tile k+1 overlaps the 4 WMMAs on tile k; one barrier per K-step.
//
// Fragment layouts (ISA 7.12.2, f16 16x16x32):
//   A 16x32: lane (h,m): elems 0..7 <-> K=8h..8h+7 ; 8..15 <-> K=16+8h..23+8h
//   B 32x16: lane (h,n): same as A, elems 0..7 <-> K=8h..8h+7 ; 8..15 <-> K=16+8h..23+8h
//   C/D f32: elem r <-> row (8h + r), col n
// ---------------------------------------------------------------------------
__global__ void __launch_bounds__(256)
wmma_gemm_kernel(const _Float16* __restrict__ A,
                 const _Float16* __restrict__ Wt,
                 const float* __restrict__ bias,
                 float* __restrict__ Cf,      // f32 out (or null)
                 _Float16* __restrict__ Ch,   // f16 out (or null)
                 int M, int Mstore, int Kd, int Nn, int act) {
  // padded LDS stride (40 halfs = 80B) keeps b128 alignment, spreads banks
  __shared__ _Float16 ldsB[2][64 * 40];

  const int stripes = Nn >> 6;
  const int rb = blockIdx.x / stripes;
  const int cs = blockIdx.x % stripes;
  const int n0 = cs << 6;
  const int wv   = threadIdx.x >> 5;
  const int lane = threadIdx.x & 31;
  const int hh = lane >> 4;   // half-wave
  const int ln = lane & 15;   // row (A) / col (B) within tile

  int m0 = rb * 128 + wv * 16;
  const bool active = (m0 < M);
  if (m0 > M - 16) m0 = M - 16;   // clamp; M is a multiple of 16, >= 16

  // cooperative B staging: thread t async-copies 16B of Wt row (n0 + t/4)
  const int bcol = threadIdx.x >> 2;          // 0..63
  const int bko  = (threadIdx.x & 3) << 3;    // 0,8,16,24
  const _Float16* wrow = Wt + (size_t)(n0 + bcol) * Kd + bko;

  // per-lane LDS byte offsets (generic LDS pointer truncates to LDS offset)
  unsigned ldsoff0 = (unsigned)(uintptr_t)&ldsB[0][bcol * 40 + bko];
  unsigned ldsoff1 = (unsigned)(uintptr_t)&ldsB[1][bcol * 40 + bko];

  const _Float16* arow = A + (size_t)(m0 + ln) * Kd + (hh << 3);

  // prologue: async-copy first B tile into buffer 0
  asm volatile("global_load_async_to_lds_b128 %0, %1, off"
               :: "v"(ldsoff0),
                  "v"((unsigned long long)(uintptr_t)wrow)
               : "memory");

  v8f acc[4] = {};
  int cur = 0;
  for (int k0 = 0; k0 < Kd; k0 += 32) {
    asm volatile("s_wait_asynccnt 0x0" ::: "memory");  // my copy landed
    __syncthreads();                                   // everyone's copy landed

    if (k0 + 32 < Kd) {   // overlap next B tile copy with this tile's WMMAs
      unsigned dsto = (cur == 0) ? ldsoff1 : ldsoff0;
      asm volatile("global_load_async_to_lds_b128 %0, %1, off"
                   :: "v"(dsto),
                      "v"((unsigned long long)(uintptr_t)(wrow + k0 + 32))
                   : "memory");
      __builtin_prefetch(arow + k0 + 32, 0, 1);
    }

    v8h alo = *(const v8h*)(arow + k0);
    v8h ahi = *(const v8h*)(arow + k0 + 16);
    v16h a = __builtin_shufflevector(alo, ahi,
                                     0, 1, 2, 3, 4, 5, 6, 7,
                                     8, 9, 10, 11, 12, 13, 14, 15);
    const _Float16* bbase = &ldsB[cur][ln * 40 + (hh << 3)];
#pragma unroll
    for (int j = 0; j < 4; ++j) {
      const _Float16* bp = bbase + j * 640;       // (16*j + ln)*40 + 8*hh
      v8h blo = *(const v8h*)(bp);
      v8h bhi = *(const v8h*)(bp + 16);
      v16h b = __builtin_shufflevector(blo, bhi,
                                       0, 1, 2, 3, 4, 5, 6, 7,
                                       8, 9, 10, 11, 12, 13, 14, 15);
      acc[j] = __builtin_amdgcn_wmma_f32_16x16x32_f16(false, a, false, b,
                                                      (short)0, acc[j],
                                                      false, false);
    }
    cur ^= 1;
  }

  if (active) {
#pragma unroll
    for (int j = 0; j < 4; ++j) {
      int col = n0 + (j << 4) + ln;
      float bs = bias ? bias[col] : 0.f;
#pragma unroll
      for (int r = 0; r < 8; ++r) {
        int row = m0 + (hh << 3) + r;
        if (row < Mstore) {
          float v = acc[j][r] + bs;
          if (act == 1) v = fmaxf(v, 0.f);                 // relu
          else if (act == 2) v = v > 0.f ? v : 0.2f * v;   // leaky_relu(0.2)
          if (Cf) Cf[(size_t)row * Nn + col] = v;
          if (Ch) Ch[(size_t)row * Nn + col] = (_Float16)v;
        }
      }
    }
  }
}

// ---------------------------------------------------------------------------
// Max over k neighbors: F[p,c] = max_k H[p,k,c]
// ---------------------------------------------------------------------------
__global__ void maxk_kernel(const _Float16* __restrict__ H,
                            _Float16* __restrict__ F, int C) {
  int gid = blockIdx.x * blockDim.x + threadIdx.x;
  if (gid >= BN_ * C) return;
  int c = gid % C;
  int p = gid / C;
  float m = -1e30f;
  for (int k = 0; k < KNN_; ++k) {
    float v = (float)H[((size_t)p * KNN_ + k) * C + c];
    m = fmaxf(m, v);
  }
  F[(size_t)p * C + c] = (_Float16)m;
}

// ---------------------------------------------------------------------------
// comb = concat([f1, f2, f3], axis=-1)   (BN x 192)
// ---------------------------------------------------------------------------
__global__ void concat3_kernel(const _Float16* __restrict__ f1,
                               const _Float16* __restrict__ f2,
                               const _Float16* __restrict__ f3,
                               _Float16* __restrict__ comb) {
  int gid = blockIdx.x * blockDim.x + threadIdx.x;
  if (gid >= BN_ * 192) return;
  int c = gid % 192;
  int p = gid / 192;
  _Float16 v;
  if (c < 64)       v = f1[(size_t)p * 64 + c];
  else if (c < 128) v = f2[(size_t)p * 64 + (c - 64)];
  else              v = f3[(size_t)p * 64 + (c - 128)];
  comb[(size_t)p * 192 + c] = v;
}

// ---------------------------------------------------------------------------
// glob[b, 0:1024] = max_n local[b,n,:], glob[b,1024:2048] = mean_n local[b,n,:]
// ---------------------------------------------------------------------------
__global__ void reduce_local_kernel(const float* __restrict__ local,
                                    float* __restrict__ glob,
                                    _Float16* __restrict__ glob16) {
  int gid = blockIdx.x * blockDim.x + threadIdx.x;
  if (gid >= B_ * 1024) return;
  int c = gid % 1024;
  int b = gid / 1024;
  const float* base = local + (size_t)b * N_ * 1024 + c;
  float mx = -1e30f, sm = 0.f;
  for (int n = 0; n < N_; ++n) {
    float v = base[(size_t)n * 1024];
    mx = fmaxf(mx, v);
    sm += v;
  }
  float mean = sm * (1.0f / (float)N_);
  glob[(size_t)b * 2048 + c]        = mx;
  glob[(size_t)b * 2048 + 1024 + c] = mean;
  glob16[(size_t)b * 2048 + c]        = (_Float16)mx;
  glob16[(size_t)b * 2048 + 1024 + c] = (_Float16)mean;
}

// ---------------------------------------------------------------------------
// Host side
// ---------------------------------------------------------------------------
static inline int nblk(long long total, int bs) {
  return (int)((total + bs - 1) / bs);
}

static inline char* carve(char*& p, size_t bytes) {
  char* r = p;
  p += (bytes + 255) & ~(size_t)255;
  return r;
}

extern "C" void kernel_launch(void* const* d_in, const int* in_sizes, int n_in,
                              void* d_out, int out_size, void* d_ws, size_t ws_size,
                              hipStream_t stream) {
  (void)in_sizes; (void)n_in; (void)out_size; (void)ws_size;

  const float* src_pc = (const float*)d_in[0];
  const float* tgt_pc = (const float*)d_in[1];
  const float* w1a = (const float*)d_in[2];  const float* b1a = (const float*)d_in[3];
  const float* w1b = (const float*)d_in[4];  const float* b1b = (const float*)d_in[5];
  const float* w2a = (const float*)d_in[6];  const float* b2a = (const float*)d_in[7];
  const float* w2b = (const float*)d_in[8];  const float* b2b = (const float*)d_in[9];
  const float* w3  = (const float*)d_in[10]; const float* b3  = (const float*)d_in[11];
  const float* w4  = (const float*)d_in[12]; const float* b4  = (const float*)d_in[13];
  const float* dw1 = (const float*)d_in[14]; const float* db1 = (const float*)d_in[15];
  const float* dw2 = (const float*)d_in[16]; const float* db2 = (const float*)d_in[17];
  const float* dw3 = (const float*)d_in[18]; const float* db3 = (const float*)d_in[19];

  float* out = (float*)d_out;
  const size_t REC_SZ   = (size_t)B_ * N_ * 3;
  const size_t LOCAL_SZ = (size_t)B_ * N_ * 1024;
  const size_t GLOB_SZ  = (size_t)B_ * 2048;
  float* out_xrec   = out;
  float* out_yrec   = out + REC_SZ;
  float* out_localx = out + 2 * REC_SZ;
  float* out_localy = out_localx + LOCAL_SZ;
  float* out_globx  = out_localy + LOCAL_SZ;
  float* out_globy  = out_globx + GLOB_SZ;

  // workspace carve (all 256B aligned). Transposed weights: Nn x Kpad.
  char* p = (char*)d_ws;
  _Float16* w1a_t = (_Float16*)carve(p, sizeof(_Float16) * 64 * 32);
  _Float16* w1b_t = (_Float16*)carve(p, sizeof(_Float16) * 64 * 64);
  _Float16* w2a_t = (_Float16*)carve(p, sizeof(_Float16) * 64 * 128);
  _Float16* w2b_t = (_Float16*)carve(p, sizeof(_Float16) * 64 * 64);
  _Float16* w3_t  = (_Float16*)carve(p, sizeof(_Float16) * 64 * 128);
  _Float16* w4_t  = (_Float16*)carve(p, sizeof(_Float16) * 1024 * 192);
  _Float16* dw1_t = (_Float16*)carve(p, sizeof(_Float16) * (size_t)2048 * 2048);
  _Float16* dw2_t = (_Float16*)carve(p, sizeof(_Float16) * (size_t)1024 * 2048);
  _Float16* dw3_t = (_Float16*)carve(p, sizeof(_Float16) * (size_t)(N_ * 3) * 1024);
  _Float16* pc16  = (_Float16*)carve(p, sizeof(_Float16) * (size_t)BN_ * 3);
  int*      idx   = (int*)     carve(p, sizeof(int) * (size_t)BN_ * KNN_);
  _Float16* gbuf  = (_Float16*)carve(p, sizeof(_Float16) * (size_t)BNK_ * 128);
  _Float16* hbuf  = (_Float16*)carve(p, sizeof(_Float16) * (size_t)BNK_ * 64);
  _Float16* f1    = (_Float16*)carve(p, sizeof(_Float16) * (size_t)BN_ * 64);
  _Float16* f2    = (_Float16*)carve(p, sizeof(_Float16) * (size_t)BN_ * 64);
  _Float16* f3    = (_Float16*)carve(p, sizeof(_Float16) * (size_t)BN_ * 64);
  _Float16* comb  = (_Float16*)carve(p, sizeof(_Float16) * (size_t)BN_ * 192);
  _Float16* glob16 = (_Float16*)carve(p, sizeof(_Float16) * 16 * 2048);  // 16 rows
  _Float16* x1    = (_Float16*)carve(p, sizeof(_Float16) * 16 * 2048);   // 16 rows
  _Float16* x2    = (_Float16*)carve(p, sizeof(_Float16) * 16 * 1024);   // 16 rows

  const int BS = 256;
  auto tcvt = [&](const float* s, _Float16* d, int K, int Nn, int Kp) {
    transpose_f32_to_f16_kernel<<<nblk((long long)Nn * Kp, BS), BS, 0, stream>>>(
        s, d, K, Nn, Kp);
  };
  auto gemm = [&](const _Float16* A, const _Float16* Wt, const float* bias,
                  float* Cf, _Float16* Ch, int M, int Mstore, int Kd, int Nn,
                  int act) {
    int blocks = ((M + 127) / 128) * (Nn / 64);
    wmma_gemm_kernel<<<blocks, 256, 0, stream>>>(A, Wt, bias, Cf, Ch,
                                                 M, Mstore, Kd, Nn, act);
  };

  // weight conversions (transpose + pad K)
  tcvt(w1a, w1a_t, 6,    64,   32);
  tcvt(w1b, w1b_t, 64,   64,   64);
  tcvt(w2a, w2a_t, 128,  64,   128);
  tcvt(w2b, w2b_t, 64,   64,   64);
  tcvt(w3,  w3_t,  128,  64,   128);
  tcvt(w4,  w4_t,  192,  1024, 192);
  tcvt(dw1, dw1_t, 2048, 2048, 2048);
  tcvt(dw2, dw2_t, 2048, 1024, 2048);
  tcvt(dw3, dw3_t, 1024, N_ * 3, 1024);

  for (int cloud = 0; cloud < 2; ++cloud) {
    const float* pc  = cloud == 0 ? src_pc : tgt_pc;
    float* out_local = cloud == 0 ? out_localx : out_localy;
    float* out_glob  = cloud == 0 ? out_globx  : out_globy;
    float* out_rec   = cloud == 0 ? out_xrec   : out_yrec;

    // ---- encoder layer 1 (g stride padded 6 -> 32) ----
    f32_to_f16_kernel<<<nblk((long long)BN_ * 3, BS), BS, 0, stream>>>(
        pc, pc16, (size_t)BN_ * 3);
    knn_kernel<3><<<nblk(BN_, BS), BS, 0, stream>>>(pc16, idx);
    edge_kernel<<<nblk((long long)BNK_ * 32, BS), BS, 0, stream>>>(pc16, idx,
                                                                   gbuf, 3, 32);
    gemm(gbuf, w1a_t, b1a, nullptr, hbuf, BNK_, BNK_, 32, 64, 1);
    gemm(hbuf, w1b_t, b1b, nullptr, gbuf, BNK_, BNK_, 64, 64, 1);
    maxk_kernel<<<nblk((long long)BN_ * 64, BS), BS, 0, stream>>>(gbuf, f1, 64);

    // ---- encoder layer 2 ----
    knn_kernel<64><<<nblk(BN_, BS), BS, 0, stream>>>(f1, idx);
    edge_kernel<<<nblk((long long)BNK_ * 128, BS), BS, 0, stream>>>(f1, idx,
                                                                    gbuf, 64, 128);
    gemm(gbuf, w2a_t, b2a, nullptr, hbuf, BNK_, BNK_, 128, 64, 1);
    gemm(hbuf, w2b_t, b2b, nullptr, gbuf, BNK_, BNK_, 64, 64, 1);
    maxk_kernel<<<nblk((long long)BN_ * 64, BS), BS, 0, stream>>>(gbuf, f2, 64);

    // ---- encoder layer 3 ----
    knn_kernel<64><<<nblk(BN_, BS), BS, 0, stream>>>(f2, idx);
    edge_kernel<<<nblk((long long)BNK_ * 128, BS), BS, 0, stream>>>(f2, idx,
                                                                    gbuf, 64, 128);
    gemm(gbuf, w3_t, b3, nullptr, hbuf, BNK_, BNK_, 128, 64, 1);
    maxk_kernel<<<nblk((long long)BN_ * 64, BS), BS, 0, stream>>>(hbuf, f3, 64);

    // ---- local features + global vector ----
    concat3_kernel<<<nblk((long long)BN_ * 192, BS), BS, 0, stream>>>(f1, f2, f3,
                                                                      comb);
    gemm(comb, w4_t, b4, out_local, nullptr, BN_, BN_, 192, 1024, 1);
    reduce_local_kernel<<<nblk(B_ * 1024, BS), BS, 0, stream>>>(out_local,
                                                                out_glob, glob16);

    // ---- decoder (A padded to 16 rows; garbage rows masked at final store) --
    gemm(glob16, dw1_t, db1, nullptr, x1, 16, 16, 2048, 2048, 2);
    gemm(x1,     dw2_t, db2, nullptr, x2, 16, 16, 2048, 1024, 2);
    gemm(x2,     dw3_t, db3, out_rec, nullptr, 16, 8, 1024, N_ * 3, 0);
  }
}
